// RefineMathPhysics_3573412790554
// MI455X (gfx1250) — matrix operand order repaired and checked by reading, not compiled, not run on hardware
//
#include <hip/hip_runtime.h>
#include <hip/hip_bf16.h>
#include <math.h>

typedef __attribute__((ext_vector_type(16))) __bf16          v16bf;
typedef __attribute__((ext_vector_type(8)))  float           v8f;

struct bfrag { uint4 lo, hi; };   // 32 bytes == one bf16 WMMA operand fragment

#define BDIM 16384
#define IND  576
#define LAT  512
#define OUTD 192
#define NSTEPS 40

// GEMM tiling: block = 256 thr = 8 waves; block tile 128x64; wave tile 32x32; K chunk 64.
#define BM 128
#define BN 64
#define BK 64
#define AST 72        // halves; 144B rows: 16B-aligned, bank-conflict-free frag gathers
#define BST 72
#define ABUF (BM * AST)   // halves per A LDS buffer
#define BBUF (BN * BST)   // halves per B LDS buffer

__device__ __forceinline__ unsigned short f2bf(float x) {
    unsigned u = __builtin_bit_cast(unsigned, x);
    unsigned r = u + 0x7FFFu + ((u >> 16) & 1u);
    return (unsigned short)(r >> 16);
}

__device__ __forceinline__ float gelu_f(float x) {
    // tanh-approx gelu; tanh(y) = 1 - 2/(exp(2y)+1)  (branch-free, v_exp_f32)
    float y  = 0.7978845608028654f * (x + 0.044715f * x * x * x);
    float th = 1.0f - 2.0f / (__expf(2.0f * y) + 1.0f);
    return 0.5f * x * (1.0f + th);
}

__device__ __forceinline__ float sigmoid_f(float x) {
    return 1.0f / (1.0f + __expf(-x));
}

// async global->LDS 16B copy; IOFFSET is added to BOTH lds and global addresses
__device__ __forceinline__ void async_cp16(unsigned lds, unsigned long long gaddr) {
    asm volatile("global_load_async_to_lds_b128 %0, %1, off" :: "v"(lds), "v"(gaddr) : "memory");
}
__device__ __forceinline__ void async_cp16_o16(unsigned lds, unsigned long long gaddr) {
    asm volatile("global_load_async_to_lds_b128 %0, %1, off offset:16" :: "v"(lds), "v"(gaddr) : "memory");
}
__device__ __forceinline__ void async_cp16_o32(unsigned lds, unsigned long long gaddr) {
    asm volatile("global_load_async_to_lds_b128 %0, %1, off offset:32" :: "v"(lds), "v"(gaddr) : "memory");
}
__device__ __forceinline__ void async_cp16_o48(unsigned lds, unsigned long long gaddr) {
    asm volatile("global_load_async_to_lds_b128 %0, %1, off offset:48" :: "v"(lds), "v"(gaddr) : "memory");
}

// ---------------------------------------------------------------------------
// C[M,N] = act( A[M,K](bf16,row-major) @ Bt[N,K](bf16,col-major of B) + bias )
// M % 128 == 0, N % 64 == 0, K % 64 == 0 (holds for every call site).
// Async global->LDS copies (ASYNCcnt) + double-buffered LDS.
// ---------------------------------------------------------------------------
template<int ACT, bool HASB1, bool WF, bool WB>
__global__ __launch_bounds__(256)
void gemm_bf16_wmma_kernel(const unsigned short* __restrict__ A,
                           const unsigned short* __restrict__ Bt,
                           const float* __restrict__ bias0,
                           const float* __restrict__ bias1, float b1scale,
                           float* __restrict__ Cf,
                           unsigned short* __restrict__ Cb,
                           int M, int N, int K)
{
    __shared__ __align__(16) unsigned short As[2 * ABUF];
    __shared__ __align__(16) unsigned short Bs[2 * BBUF];

    const int tid  = threadIdx.x;
    const int lane = tid & 31;
    const int wave = tid >> 5;
    const int wm   = (wave & 3) * 32;   // wave row offset within block tile
    const int wn   = (wave >> 2) * 32;  // wave col offset within block tile
    const int m0   = blockIdx.y * BM;
    const int n0   = blockIdx.x * BN;

    // cooperative copy shares: A 128x64 halves (32/thr), B 64x64 halves (16/thr)
    const int ar = tid >> 1, ac = (tid & 1) * 32;
    const int br = tid >> 2, bc = (tid & 3) * 16;

    const unsigned short* Ag = A  + (size_t)(m0 + ar) * K + ac;
    const unsigned short* Bg = Bt + (size_t)(n0 + br) * K + bc;

    const unsigned la_st = (unsigned)(uintptr_t)(void*)&As[0] + (unsigned)((ar * AST + ac) * 2);
    const unsigned lb_st = (unsigned)(uintptr_t)(void*)&Bs[0] + (unsigned)((br * BST + bc) * 2);

    auto issue_copy = [&](int buf, int it) {
        const unsigned la = la_st + (unsigned)(buf * ABUF * 2);
        const unsigned lb = lb_st + (unsigned)(buf * BBUF * 2);
        const unsigned long long ga = (unsigned long long)(uintptr_t)(Ag + it * BK);
        const unsigned long long gb = (unsigned long long)(uintptr_t)(Bg + it * BK);
        async_cp16    (la, ga);
        async_cp16_o16(la, ga);
        async_cp16_o32(la, ga);
        async_cp16_o48(la, ga);
        async_cp16    (lb, gb);
        async_cp16_o16(lb, gb);
    };

    // per-lane fragment indices (buffer-invariant)
    const int a0i = (wm +      (lane & 15)) * AST;
    const int a1i = (wm + 16 + (lane & 15)) * AST;
    const int b0i = (wn +      (lane & 15)) * BST;
    const int b1i = (wn + 16 + (lane & 15)) * BST;
    const int kha = (lane >> 4) * 8;
    const int khb = (lane >> 4) * 8;

    issue_copy(0, 0);
    asm volatile("s_wait_asynccnt 0" ::: "memory");
    __syncthreads();

    v8f c00 = {}, c01 = {}, c10 = {}, c11 = {};
    const int nk = K / BK;

    for (int it = 0; it < nk; ++it) {
        const int  cur  = it & 1;
        const bool more = (it + 1) < nk;
        if (more) issue_copy(cur ^ 1, it + 1);   // async; overlapped with WMMAs below

        const unsigned short* Ab = &As[cur * ABUF];
        const unsigned short* Bb = &Bs[cur * BBUF];
        #pragma unroll
        for (int kc = 0; kc < 2; ++kc) {
            const int ka = kc * 32 + kha;
            const int kb = kc * 32 + khb;
            v16bf a0 = __builtin_bit_cast(v16bf,
                bfrag{ *(const uint4*)(Ab + a0i + ka), *(const uint4*)(Ab + a0i + ka + 16) });
            v16bf a1 = __builtin_bit_cast(v16bf,
                bfrag{ *(const uint4*)(Ab + a1i + ka), *(const uint4*)(Ab + a1i + ka + 16) });
            v16bf b0 = __builtin_bit_cast(v16bf,
                bfrag{ *(const uint4*)(Bb + b0i + kb), *(const uint4*)(Bb + b0i + kb + 16) });
            v16bf b1 = __builtin_bit_cast(v16bf,
                bfrag{ *(const uint4*)(Bb + b1i + kb), *(const uint4*)(Bb + b1i + kb + 16) });
            c00 = __builtin_amdgcn_wmma_f32_16x16x32_bf16(false, a0, false, b0, (short)0, c00, false, false);
            c01 = __builtin_amdgcn_wmma_f32_16x16x32_bf16(false, a0, false, b1, (short)0, c01, false, false);
            c10 = __builtin_amdgcn_wmma_f32_16x16x32_bf16(false, a1, false, b0, (short)0, c10, false, false);
            c11 = __builtin_amdgcn_wmma_f32_16x16x32_bf16(false, a1, false, b1, (short)0, c11, false, false);
        }

        if (more) asm volatile("s_wait_asynccnt 0" ::: "memory");
        __syncthreads();
    }

    // C/D layout: lanes 0-15 -> M=r, N=lane; lanes 16-31 -> M=8+r, N=lane-16
    v8f accs[2][2] = { { c00, c01 }, { c10, c11 } };
    #pragma unroll
    for (int ti = 0; ti < 2; ++ti) {
        const int rowb = m0 + wm + ti * 16 + (lane >> 4) * 8;
        #pragma unroll
        for (int tj = 0; tj < 2; ++tj) {
            const int col = n0 + wn + tj * 16 + (lane & 15);
            float bsum = bias0[col];
            if (HASB1) bsum += b1scale * bias1[col];
            #pragma unroll
            for (int r = 0; r < 8; ++r) {
                float x = accs[ti][tj][r] + bsum;
                if (ACT) x = gelu_f(x);
                const size_t idx = (size_t)(rowb + r) * N + col;
                if (WF) Cf[idx] = x;
                if (WB) Cb[idx] = f2bf(x);
            }
        }
    }
}

// ---------------------------------------------------------------------------
__device__ __forceinline__ float block_reduce_256(float val, float* red) {
    #pragma unroll
    for (int o = 16; o > 0; o >>= 1) val += __shfl_xor(val, o, 32);
    if ((threadIdx.x & 31) == 0) red[threadIdx.x >> 5] = val;
    __syncthreads();
    float r = 0.f;
    #pragma unroll
    for (int i = 0; i < 8; ++i) r += red[i];
    __syncthreads();
    return r;
}

// one block (256 thr) per row: velocity, LayerNorm, halt head, accumulation
__global__ __launch_bounds__(256)
void row_update_kernel(float* __restrict__ z, unsigned short* __restrict__ zb,
                       float* __restrict__ vel, const float* __restrict__ force,
                       const float* __restrict__ ln_scale, const float* __restrict__ ln_bias,
                       const float* __restrict__ halt_w, const float* __restrict__ halt_b,
                       float* __restrict__ runp, float* __restrict__ wz,
                       float* __restrict__ step_probs, int step_off)
{
    __shared__ float red[8];
    __shared__ float pbc;
    const int row = blockIdx.x;
    const size_t base = (size_t)row * LAT;
    const int c0 = threadIdx.x, c1 = threadIdx.x + 256;

    float v0 = 0.6f * vel[base + c0] + 0.1f * force[base + c0];
    float v1 = 0.6f * vel[base + c1] + 0.1f * force[base + c1];
    vel[base + c0] = v0; vel[base + c1] = v1;
    float t0 = z[base + c0] + v0, t1 = z[base + c1] + v1;

    float s1 = block_reduce_256(t0 + t1, red);
    float s2 = block_reduce_256(t0 * t0 + t1 * t1, red);
    float mean = s1 * (1.0f / (float)LAT);
    float var  = s2 * (1.0f / (float)LAT) - mean * mean;
    float rstd = rsqrtf(var + 1e-6f);
    float nz0 = (t0 - mean) * rstd * ln_scale[c0] + ln_bias[c0];
    float nz1 = (t1 - mean) * rstd * ln_scale[c1] + ln_bias[c1];

    float hd = block_reduce_256(nz0 * halt_w[c0] + nz1 * halt_w[c1], red);
    if (threadIdx.x == 0) {
        float halt = sigmoid_f(hd + halt_b[0]);
        float rp = runp[row];
        float p = halt * (1.0f - rp);
        runp[row] = rp + p;
        step_probs[step_off + row] = p;
        pbc = p;
    }
    __syncthreads();
    float p = pbc;

    z[base + c0] = nz0;        z[base + c1] = nz1;
    zb[base + c0] = f2bf(nz0); zb[base + c1] = f2bf(nz1);
    wz[base + c0] += p * nz0;  wz[base + c1] += p * nz1;
}

// predicted_steps = sigmoid(z @ comp_w + comp_b) * max_steps
__global__ __launch_bounds__(256)
void comp_kernel(const float* __restrict__ z, const float* __restrict__ comp_w,
                 const float* __restrict__ comp_b, const int* __restrict__ max_steps,
                 float* __restrict__ out)
{
    __shared__ float red[8];
    const int row = blockIdx.x;
    const size_t base = (size_t)row * LAT;
    float d = z[base + threadIdx.x]       * comp_w[threadIdx.x] +
              z[base + threadIdx.x + 256] * comp_w[threadIdx.x + 256];
    float s = block_reduce_256(d, red);
    if (threadIdx.x == 0)
        out[row] = sigmoid_f(s + comp_b[0]) * (float)(*max_steps);
}

__global__ void finalize_kernel(const float* __restrict__ z, const float* __restrict__ runp,
                                float* __restrict__ wz, unsigned short* __restrict__ wzb)
{
    const size_t i = (size_t)blockIdx.x * blockDim.x + threadIdx.x; // B*LAT
    const int row = (int)(i >> 9);
    float rem = 1.0f - runp[row];
    float w = wz[i] + rem * z[i];
    wz[i] = w;
    wzb[i] = f2bf(w);
}

__global__ void cvt_f32_bf16_kernel(const float* __restrict__ in,
                                    unsigned short* __restrict__ out, int n)
{
    const int i = blockIdx.x * blockDim.x + threadIdx.x;
    if (i < n) out[i] = f2bf(in[i]);
}

// transpose + convert weights: in [K,N] f32 row-major -> out [N,K] bf16 (B^T)
__global__ void wtrans_bf16_kernel(const float* __restrict__ in,
                                   unsigned short* __restrict__ out, int K, int N)
{
    const int i = blockIdx.x * blockDim.x + threadIdx.x;
    if (i < K * N) {
        const int k = i / N, n = i - k * N;
        out[(size_t)n * K + k] = f2bf(in[i]);
    }
}

__global__ void zero_f32_kernel(float* __restrict__ p, size_t n)
{
    const size_t i = (size_t)blockIdx.x * blockDim.x + threadIdx.x;
    if (i < n) p[i] = 0.0f;
}

// ---------------------------------------------------------------------------
extern "C" void kernel_launch(void* const* d_in, const int* in_sizes, int n_in,
                              void* d_out, int out_size, void* d_ws, size_t ws_size,
                              hipStream_t stream)
{
    (void)in_sizes; (void)n_in; (void)out_size; (void)ws_size;

    const float* raw      = (const float*)d_in[0];
    const float* enc_w    = (const float*)d_in[1];
    const float* enc_b    = (const float*)d_in[2];
    const float* dec_w    = (const float*)d_in[3];
    const float* dec_b    = (const float*)d_in[4];
    const float* rec_w    = (const float*)d_in[5];
    const float* rec_b    = (const float*)d_in[6];
    const float* fc1_w    = (const float*)d_in[7];
    const float* fc1_b    = (const float*)d_in[8];
    const float* fc2_w    = (const float*)d_in[9];
    const float* fc2_b    = (const float*)d_in[10];
    const float* ln_scale = (const float*)d_in[11];
    const float* ln_bias  = (const float*)d_in[12];
    const float* halt_w   = (const float*)d_in[13];
    const float* halt_b   = (const float*)d_in[14];
    const float* comp_w   = (const float*)d_in[15];
    const float* comp_b   = (const float*)d_in[16];
    const int*   msteps   = (const int*)d_in[17];

    float* out    = (float*)d_out;
    float* w_out  = out;                                  // [B,192]
    float* w_z    = w_out + (size_t)BDIM * OUTD;          // [B,512]
    float* recog  = w_z   + (size_t)BDIM * LAT;           // [B,576]
    float* sprob  = recog + (size_t)BDIM * IND;           // [40,B,1]
    float* psteps = sprob + (size_t)NSTEPS * BDIM;        // [B,1]

    char* ws = (char*)d_ws;
    size_t off = 0;
    auto take = [&](size_t bytes) -> void* {
        void* p = ws + off;
        off = (off + bytes + 255) & ~(size_t)255;
        return p;
    };
    unsigned short* x_bf  = (unsigned short*)take((size_t)BDIM * IND * 2);
    unsigned short* encwT = (unsigned short*)take((size_t)IND * LAT * 2);   // [512,576]
    unsigned short* fc1wT = (unsigned short*)take((size_t)LAT * 1024 * 2);  // [1024,512]
    unsigned short* fc2wT = (unsigned short*)take((size_t)1024 * LAT * 2);  // [512,1024]
    unsigned short* decwT = (unsigned short*)take((size_t)LAT * OUTD * 2);  // [192,512]
    unsigned short* recwT = (unsigned short*)take((size_t)LAT * IND * 2);   // [576,512]
    float*          zf    = (float*)take((size_t)BDIM * LAT * 4);
    unsigned short* zb    = (unsigned short*)take((size_t)BDIM * LAT * 2);
    float*          vel   = (float*)take((size_t)BDIM * LAT * 4);
    unsigned short* hb    = (unsigned short*)take((size_t)BDIM * 1024 * 2);
    float*          force = (float*)take((size_t)BDIM * LAT * 4);
    float*          runp  = (float*)take((size_t)BDIM * 4);
    unsigned short* wzb   = (unsigned short*)take((size_t)BDIM * LAT * 2);

    cvt_f32_bf16_kernel<<<(BDIM * IND + 255) / 256, 256, 0, stream>>>(raw, x_bf, BDIM * IND);
    auto wtrans = [&](const float* src, unsigned short* dst, int K, int N) {
        wtrans_bf16_kernel<<<(K * N + 255) / 256, 256, 0, stream>>>(src, dst, K, N);
    };
    wtrans(enc_w, encwT, IND, LAT);
    wtrans(fc1_w, fc1wT, LAT, 1024);      // rows 0..511; row 512 folded into step bias
    wtrans(fc2_w, fc2wT, 1024, LAT);
    wtrans(dec_w, decwT, LAT, OUTD);
    wtrans(rec_w, recwT, LAT, IND);

    zero_f32_kernel<<<(BDIM * LAT + 255) / 256, 256, 0, stream>>>(vel, (size_t)BDIM * LAT);
    zero_f32_kernel<<<(BDIM + 255) / 256, 256, 0, stream>>>(runp, (size_t)BDIM);
    zero_f32_kernel<<<(BDIM * LAT + 255) / 256, 256, 0, stream>>>(w_z, (size_t)BDIM * LAT);

    // encoder: z = gelu(x @ enc_w + enc_b)   (f32 + bf16 outputs)
    gemm_bf16_wmma_kernel<1, false, true, true>
        <<<dim3(LAT / BN, BDIM / BM), 256, 0, stream>>>(
        x_bf, encwT, enc_b, nullptr, 0.0f, zf, zb, BDIM, LAT, IND);

    comp_kernel<<<BDIM, 256, 0, stream>>>(zf, comp_w, comp_b, msteps, psteps);

    const float* fc1_row512 = fc1_w + (size_t)LAT * 1024; // step-feature row of fc1_w
    for (int s = 0; s < NSTEPS; ++s) {
        // h = gelu(z @ fc1_w[:512] + fc1_b + s * fc1_w[512])   (bf16 output only)
        gemm_bf16_wmma_kernel<1, true, false, true>
            <<<dim3(1024 / BN, BDIM / BM), 256, 0, stream>>>(
            zb, fc1wT, fc1_b, fc1_row512, (float)s, nullptr, hb, BDIM, 1024, LAT);
        // force = h @ fc2_w + fc2_b   (f32 output only)
        gemm_bf16_wmma_kernel<0, false, true, false>
            <<<dim3(LAT / BN, BDIM / BM), 256, 0, stream>>>(
            hb, fc2wT, fc2_b, nullptr, 0.0f, force, nullptr, BDIM, LAT, 1024);
        row_update_kernel<<<BDIM, 256, 0, stream>>>(
            zf, zb, vel, force, ln_scale, ln_bias, halt_w, halt_b,
            runp, w_z, sprob, s * BDIM);
    }

    finalize_kernel<<<(BDIM * LAT) / 256, 256, 0, stream>>>(zf, runp, w_z, wzb);

    // w_out = w_z @ dec_w + dec_b  (all per-step decoder GEMMs folded here)
    gemm_bf16_wmma_kernel<0, false, true, false>
        <<<dim3(OUTD / BN, BDIM / BM), 256, 0, stream>>>(
        wzb, decwT, dec_b, nullptr, 0.0f, w_out, nullptr, BDIM, OUTD, LAT);
    // recog = w_z @ rec_w + rec_b
    gemm_bf16_wmma_kernel<0, false, true, false>
        <<<dim3(IND / BN, BDIM / BM), 256, 0, stream>>>(
        wzb, recwT, rec_b, nullptr, 0.0f, recog, nullptr, BDIM, IND, LAT);
}
